// MambaEncoder_29008209117723
// MI455X (gfx1250) — compile-verified, matrix operand order checked by reading, not run on hardware
//
#include <hip/hip_runtime.h>
#include <math.h>

// ---------------------------------------------------------------------------
// MambaEncoder for MI455X (gfx1250, wave32, WMMA).
// Heavy GEMMs run on v_wmma_f32_16x16x32_bf16. fp32->bf16 conversion is done
// during LDS staging with one v_perm_b32 per packed pair (round-half-up via
// +0x8000), and the K-loop is software-pipelined with ping-pong LDS tiles
// (one barrier per K-step, global loads overlap WMMA issue).
// ---------------------------------------------------------------------------

#define B__  8
#define L__  2048
#define F__  512
#define DI__ 1024
#define NST  16
#define DTR  32
#define M__  (B__ * L__)   // 16384 tokens

#define BM 128
#define BN 128
#define BK 32
#define SA 40   // LDS row stride in ushorts: 80B = 20 banks -> conflict-free

typedef __attribute__((ext_vector_type(16))) __bf16 v16bf;
typedef __attribute__((ext_vector_type(8)))  float  v8f;

union FragU { v16bf v; unsigned int u[8]; };

// Pack two fp32 -> two bf16 (round-half-up) in one v_perm_b32.
// dst[15:0] = (lo+0x8000)[31:16], dst[31:16] = (hi+0x8000)[31:16]
__device__ __forceinline__ unsigned int pack_bf16x2(float lo, float hi) {
  const unsigned int ulo = __float_as_uint(lo) + 0x8000u;
  const unsigned int uhi = __float_as_uint(hi) + 0x8000u;
  return __builtin_amdgcn_perm(uhi, ulo, 0x07060302u);
}

// A/B fragment gather from an LDS tile laid out [row][k] (bf16, stride SA).
// 16-bit operand layout (wave32): lane holds row/col = lane&15; half=lane>>4;
// VGPR v holds K = (v>>2)*16 + half*8 + (v&3)*2 .. +1 (pairs contiguous ->
// the 8 dword reads collapse into two ds_load_b128).
__device__ __forceinline__ v16bf frag_from_lds(const unsigned short* lds,
                                               int rowBase, int lane) {
  const int r = rowBase + (lane & 15);
  const int half = (lane >> 4) & 1;
  const unsigned int* p = (const unsigned int*)(lds + r * SA);
  FragU f;
#pragma unroll
  for (int v = 0; v < 8; ++v) {
    const int k = ((v >> 2) << 4) + (half << 3) + ((v & 3) << 1);
    f.u[v] = p[k >> 1];
  }
  return f.v;
}

__device__ __forceinline__ float apply_act(float v, int act) {
  if (act == 1) return fmaxf(v, 0.0f);                         // relu
  if (act == 2) return (v > 20.0f) ? v : log1pf(__expf(v));    // softplus
  return v;
}

// Stage one 128x32 A tile + 128x32 W tile (fp32 global -> bf16 LDS).
__device__ __forceinline__ void stage_ab(
    unsigned short* __restrict__ sA, unsigned short* __restrict__ sW,
    const float* __restrict__ A, int lda,
    const float* __restrict__ W, int ldw,
    int m0, int n0, int N, int kt, int r, int c0)
{
  {
    const float* src = A + (size_t)(m0 + r) * lda + kt + c0;
    unsigned int* dst = (unsigned int*)(sA + r * SA + c0);
#pragma unroll
    for (int j = 0; j < 4; ++j) {
      float4 f = ((const float4*)src)[j];
      dst[2*j+0] = pack_bf16x2(f.x, f.y);
      dst[2*j+1] = pack_bf16x2(f.z, f.w);
    }
  }
  {
    const int n = n0 + r;
    unsigned int* dst = (unsigned int*)(sW + r * SA + c0);
    if (n < N) {
      const float* src = W + (size_t)n * ldw + kt + c0;
#pragma unroll
      for (int j = 0; j < 4; ++j) {
        float4 f = ((const float4*)src)[j];
        dst[2*j+0] = pack_bf16x2(f.x, f.y);
        dst[2*j+1] = pack_bf16x2(f.z, f.w);
      }
    } else {
#pragma unroll
      for (int j = 0; j < 8; ++j) dst[j] = 0u;
    }
  }
}

// ---------------------------------------------------------------------------
// Generic GEMM: Out[m,n] = act( sum_k A[m,k]*W[n,k] + bias[n] ) + residual
// Block: 256 thr = 8 waves; tile 128x128; each wave computes 32x64 (2x4 frags).
// Ping-pong LDS double buffering: one barrier per K-step.
// M % 128 == 0 (always 16384 here); N guarded; K % 32 == 0.
// ---------------------------------------------------------------------------
__global__ __launch_bounds__(256) void gemm_wmma(
    const float* __restrict__ A, int lda,
    const float* __restrict__ W, int ldw,
    float* __restrict__ Out, int ldo,
    const float* __restrict__ bias,
    const float* __restrict__ residual,
    int M, int N, int K, int act)
{
  __shared__ unsigned short As[2][BM * SA];
  __shared__ unsigned short Ws[2][BN * SA];

  const int tid  = threadIdx.x;
  const int lane = tid & 31;
  const int wave = tid >> 5;
  const int m0 = blockIdx.y * BM;
  const int n0 = blockIdx.x * BN;
  const int wr = wave & 3;   // 4 waves along M -> rows wr*32..+31
  const int wc = wave >> 2;  // 2 waves along N -> cols wc*64..+63

  v8f acc[2][4];
#pragma unroll
  for (int i = 0; i < 2; ++i)
#pragma unroll
    for (int j = 0; j < 4; ++j) { v8f z = {}; acc[i][j] = z; }

  const int r  = tid >> 1;          // 0..127 (tile row)
  const int c0 = (tid & 1) << 4;    // 0 or 16 (k segment)

  stage_ab(As[0], Ws[0], A, lda, W, ldw, m0, n0, N, 0, r, c0);
  __syncthreads();

  int p = 0;
  for (int kt = 0; kt < K; kt += BK) {
    if (kt + 2 * BK < K)  // prefetch A stream two steps ahead -> global_prefetch
      __builtin_prefetch(A + (size_t)(m0 + r) * lda + kt + 2 * BK + c0, 0, 1);
    if (kt + BK < K)
      stage_ab(As[p ^ 1], Ws[p ^ 1], A, lda, W, ldw, m0, n0, N, kt + BK, r, c0);

    const v16bf a0 = frag_from_lds(As[p], wr * 32 + 0,  lane);
    const v16bf a1 = frag_from_lds(As[p], wr * 32 + 16, lane);
#pragma unroll
    for (int j = 0; j < 4; ++j) {
      const v16bf b = frag_from_lds(Ws[p], wc * 64 + j * 16, lane);
      acc[0][j] = __builtin_amdgcn_wmma_f32_16x16x32_bf16(
          false, a0, false, b, (short)0, acc[0][j], false, false);
      acc[1][j] = __builtin_amdgcn_wmma_f32_16x16x32_bf16(
          false, a1, false, b, (short)0, acc[1][j], false, false);
    }
    __syncthreads();
    p ^= 1;
  }

  // ---- epilogue: D layout lanes0-15 n=lane, VGPR v -> m = half*8 + v ----
  const int half  = (lane >> 4) & 1;
  const int nlane = lane & 15;
#pragma unroll
  for (int i = 0; i < 2; ++i) {
#pragma unroll
    for (int j = 0; j < 4; ++j) {
      const int n = n0 + wc * 64 + j * 16 + nlane;
      if (n >= N) continue;
      const float bv = bias ? bias[n] : 0.0f;
#pragma unroll
      for (int v = 0; v < 8; ++v) {
        const int m = m0 + wr * 32 + i * 16 + half * 8 + v;
        float val = apply_act(acc[i][j][v] + bv, act);
        if (residual) val += residual[(size_t)m * ldo + n];
        Out[(size_t)m * ldo + n] = val;
      }
    }
  }
}

// ---------------------------------------------------------------------------
// Full "same" conv1d as WMMA GEMM: Out[b,l,co] = bias[co] +
//   sum_dk sum_ci X[b, l+dk-ks/2, ci] * Wc[co, ci, dk]
// (dk, kt) loops flattened so the same ping-pong pipeline applies.
// ---------------------------------------------------------------------------
__device__ __forceinline__ void stage_conv(
    unsigned short* __restrict__ sA, unsigned short* __restrict__ sW,
    const float* __restrict__ X, const float* __restrict__ Wc,
    int Fdim, int ks, int m0, int n0, int dk, int kt, int r, int c0)
{
  const int m  = m0 + r;
  const int bb = m >> 11;            // / L__
  const int ll = m & (L__ - 1);
  const int lp = ll + dk - ks / 2;
  unsigned int* dstA = (unsigned int*)(sA + r * SA + c0);
  if (lp >= 0 && lp < L__) {
    const float* src = X + ((size_t)bb * L__ + lp) * Fdim + kt + c0;
#pragma unroll
    for (int j = 0; j < 4; ++j) {
      float4 f = ((const float4*)src)[j];
      dstA[2*j+0] = pack_bf16x2(f.x, f.y);
      dstA[2*j+1] = pack_bf16x2(f.z, f.w);
    }
  } else {
#pragma unroll
    for (int j = 0; j < 8; ++j) dstA[j] = 0u;
  }
  const int n = n0 + r;
  const float* wbase = Wc + (size_t)n * Fdim * ks + dk;   // stride ks over ci
  unsigned int* dstW = (unsigned int*)(sW + r * SA + c0);
#pragma unroll
  for (int e = 0; e < 8; ++e) {
    const float lo = wbase[(size_t)(kt + c0 + 2*e + 0) * ks];
    const float hi = wbase[(size_t)(kt + c0 + 2*e + 1) * ks];
    dstW[e] = pack_bf16x2(lo, hi);
  }
}

__global__ __launch_bounds__(256) void conv_gemm_wmma(
    const float* __restrict__ X,   // [B, L, F]
    const float* __restrict__ Wc,  // [F, F, ks]
    const float* __restrict__ bias,
    float* __restrict__ Out,       // [B, L, F]
    int Fdim, int ks)
{
  __shared__ unsigned short As[2][BM * SA];
  __shared__ unsigned short Ws[2][BN * SA];

  const int tid  = threadIdx.x;
  const int lane = tid & 31;
  const int wave = tid >> 5;
  const int m0 = blockIdx.y * BM;
  const int n0 = blockIdx.x * BN;
  const int wr = wave & 3;
  const int wc = wave >> 2;

  v8f acc[2][4];
#pragma unroll
  for (int i = 0; i < 2; ++i)
#pragma unroll
    for (int j = 0; j < 4; ++j) { v8f z = {}; acc[i][j] = z; }

  const int r  = tid >> 1;
  const int c0 = (tid & 1) << 4;

  const int ksteps = Fdim / BK;       // K-steps per tap
  const int total  = ks * ksteps;

  stage_conv(As[0], Ws[0], X, Wc, Fdim, ks, m0, n0, 0, 0, r, c0);
  __syncthreads();

  int p = 0;
  for (int s = 0; s < total; ++s) {
    if (s + 1 < total) {
      const int dk = (s + 1) / ksteps;
      const int kt = ((s + 1) % ksteps) * BK;
      stage_conv(As[p ^ 1], Ws[p ^ 1], X, Wc, Fdim, ks, m0, n0, dk, kt, r, c0);
    }
    const v16bf a0 = frag_from_lds(As[p], wr * 32 + 0,  lane);
    const v16bf a1 = frag_from_lds(As[p], wr * 32 + 16, lane);
#pragma unroll
    for (int j = 0; j < 4; ++j) {
      const v16bf b = frag_from_lds(Ws[p], wc * 64 + j * 16, lane);
      acc[0][j] = __builtin_amdgcn_wmma_f32_16x16x32_bf16(
          false, a0, false, b, (short)0, acc[0][j], false, false);
      acc[1][j] = __builtin_amdgcn_wmma_f32_16x16x32_bf16(
          false, a1, false, b, (short)0, acc[1][j], false, false);
    }
    __syncthreads();
    p ^= 1;
  }

  const int half  = (lane >> 4) & 1;
  const int nlane = lane & 15;
#pragma unroll
  for (int i = 0; i < 2; ++i) {
#pragma unroll
    for (int j = 0; j < 4; ++j) {
      const int n = n0 + wc * 64 + j * 16 + nlane;
      const float bv = bias[n];
#pragma unroll
      for (int v = 0; v < 8; ++v) {
        const int mm = m0 + wr * 32 + i * 16 + half * 8 + v;
        Out[(size_t)mm * Fdim + n] = acc[i][j][v] + bv;
      }
    }
  }
}

// ---------------------------------------------------------------------------
// Depthwise causal conv (k=4, left pad 3) + bias + SiLU.
// Input rows live in xz[:, :DI] (row stride 2*DI); output xm (stride DI).
// ---------------------------------------------------------------------------
__global__ __launch_bounds__(256) void dwconv_silu(
    const float* __restrict__ xz, const float* __restrict__ cw,
    const float* __restrict__ cb, float* __restrict__ xm)
{
  const int idx = blockIdx.x * 256 + threadIdx.x;   // over M*DI
  const int d = idx & (DI__ - 1);
  const int t = idx >> 10;
  const int l = t & (L__ - 1);
  float acc = cb[d];
#pragma unroll
  for (int j = 0; j < 4; ++j) {
    const int lp = l - 3 + j;
    if (lp >= 0)
      acc = fmaf(xz[(size_t)(t - 3 + j) * (2 * DI__) + d], cw[d * 4 + j], acc);
  }
  xm[(size_t)t * DI__ + d] = acc / (1.0f + __expf(-acc));
}

// ---------------------------------------------------------------------------
// Selective scan fused with y*silu(z) gating. One thread per (b,d) channel,
// 16 states in registers; B/C loads are block-uniform -> broadcast.
// out may alias u (same-index read-then-write).
// ---------------------------------------------------------------------------
__global__ __launch_bounds__(256) void sscan_k(
    const float* __restrict__ u,      // [B*L, DI]
    const float* __restrict__ delta,  // [B*L, 2*DI] (aliased into xz x-half)
    const float* __restrict__ xdbc,   // [B*L, 64]; B at +32, C at +48
    const float* __restrict__ z,      // [B*L, 2*DI] (xz + DI)
    const float* __restrict__ A_log,  // [DI, 16]
    const float* __restrict__ Dp,     // [DI]
    float* __restrict__ out)          // [B*L, DI]
{
  const int g = blockIdx.x * 256 + threadIdx.x;
  const int b = g >> 10;
  const int d = g & (DI__ - 1);
  float Arow[NST];
#pragma unroll
  for (int n = 0; n < NST; ++n) Arow[n] = -__expf(A_log[d * NST + n]);
  const float Dd = Dp[d];
  float h[NST];
#pragma unroll
  for (int n = 0; n < NST; ++n) h[n] = 0.0f;

  size_t tok = (size_t)b * L__;
  for (int l = 0; l < L__; ++l, ++tok) {
    const float dt = delta[tok * (2 * DI__) + d];
    const float ut = u[tok * DI__ + d];
    const float zt = z[tok * (2 * DI__) + d];
    const float du = dt * ut;
    const float* bc = xdbc + tok * (DTR + 2 * NST);
    float y = 0.0f;
#pragma unroll
    for (int n = 0; n < NST; ++n) {
      const float Bn = bc[DTR + n];
      const float Cn = bc[DTR + NST + n];
      h[n] = fmaf(h[n], __expf(dt * Arow[n]), du * Bn);
      y = fmaf(h[n], Cn, y);
    }
    y = fmaf(ut, Dd, y);
    const float sz = zt / (1.0f + __expf(-zt));
    out[tok * DI__ + d] = y * sz;
  }
}

// ---------------------------------------------------------------------------
// RMSNorm: one wave32 per token (F=512 -> 16 elements/lane, shfl reduction).
// ---------------------------------------------------------------------------
__global__ __launch_bounds__(256) void rmsnorm_k(
    const float* __restrict__ x, const float* __restrict__ w,
    float* __restrict__ y)
{
  const int lane = threadIdx.x & 31;
  const int tok  = blockIdx.x * 8 + (threadIdx.x >> 5);
  const float* xr = x + (size_t)tok * F__;
  float vals[16];
  float ss = 0.0f;
#pragma unroll
  for (int i = 0; i < 16; ++i) { vals[i] = xr[lane + 32 * i]; ss = fmaf(vals[i], vals[i], ss); }
#pragma unroll
  for (int m = 16; m >= 1; m >>= 1) ss += __shfl_xor(ss, m, 32);
  const float inv = rsqrtf(ss * (1.0f / F__) + 1e-5f);
#pragma unroll
  for (int i = 0; i < 16; ++i) {
    const int c = lane + 32 * i;
    y[(size_t)tok * F__ + c] = vals[i] * inv * w[c];
  }
}

// ---------------------------------------------------------------------------
// LayerNorm (optionally + addsrc, fuses the final "+ input_seq").
// ---------------------------------------------------------------------------
__global__ __launch_bounds__(256) void layernorm_k(
    const float* __restrict__ x, const float* __restrict__ w,
    const float* __restrict__ b, const float* __restrict__ addsrc,
    float* __restrict__ y)
{
  const int lane = threadIdx.x & 31;
  const int tok  = blockIdx.x * 8 + (threadIdx.x >> 5);
  const float* xr = x + (size_t)tok * F__;
  float vals[16];
  float s = 0.0f;
#pragma unroll
  for (int i = 0; i < 16; ++i) { vals[i] = xr[lane + 32 * i]; s += vals[i]; }
#pragma unroll
  for (int m = 16; m >= 1; m >>= 1) s += __shfl_xor(s, m, 32);
  const float mu = s * (1.0f / F__);
  float var = 0.0f;
#pragma unroll
  for (int i = 0; i < 16; ++i) { const float dv = vals[i] - mu; var = fmaf(dv, dv, var); }
#pragma unroll
  for (int m = 16; m >= 1; m >>= 1) var += __shfl_xor(var, m, 32);
  const float inv = rsqrtf(var * (1.0f / F__) + 1e-5f);
#pragma unroll
  for (int i = 0; i < 16; ++i) {
    const int c = lane + 32 * i;
    float o = (vals[i] - mu) * inv * w[c] + b[c];
    if (addsrc) o += addsrc[(size_t)tok * F__ + c];
    y[(size_t)tok * F__ + c] = o;
  }
}

// ---------------------------------------------------------------------------
// Host-side orchestration. Input flattening assumed jax-tree order
// (dicts by sorted key, lists in order):
//   0: input_seq
//   1,3,5: cnn[k].b (k=3,5,7)   2,4,6: cnn[k].w
//   7: final_ln_b  8: final_ln_w
//   per layer (base = 9 + 16*layer): A_log, D, conv_b, conv_w, dt_b, dt_w,
//   fc1_b, fc1_w, fc2_b, fc2_w, in_proj, ln_b, ln_w, out_proj, rms_w, x_proj
// ---------------------------------------------------------------------------
extern "C" void kernel_launch(void* const* d_in, const int* in_sizes, int n_in,
                              void* d_out, int out_size, void* d_ws, size_t ws_size,
                              hipStream_t stream) {
  (void)in_sizes; (void)n_in; (void)out_size; (void)ws_size;
  const float* x_in  = (const float*)d_in[0];
  const float* fln_b = (const float*)d_in[7];
  const float* fln_w = (const float*)d_in[8];

  float* ws       = (float*)d_ws;
  float* buf_h    = ws;                                   // [M, 512]
  float* buf_t    = buf_h  + (size_t)M__ * F__;           // [M, 512]
  float* buf_xz   = buf_t  + (size_t)M__ * F__;           // [M, 2048]
  float* buf_xm   = buf_xz + (size_t)M__ * 2 * DI__;      // [M, 1024]
  float* buf_xdbc = buf_xm + (size_t)M__ * DI__;          // [M, 64]
  // total: 16384 * 4160 floats = 272.6 MB

  const dim3 blk(256);
  const dim3 gConv(F__ / BN, M__ / BM);                   // (4, 128)

  // CNN front-end (full convs as shifted WMMA GEMMs)
  conv_gemm_wmma<<<gConv, blk, 0, stream>>>(x_in,  (const float*)d_in[2],
                                            (const float*)d_in[1], buf_h, F__, 3);
  conv_gemm_wmma<<<gConv, blk, 0, stream>>>(buf_h, (const float*)d_in[4],
                                            (const float*)d_in[3], buf_t, F__, 5);
  conv_gemm_wmma<<<gConv, blk, 0, stream>>>(buf_t, (const float*)d_in[6],
                                            (const float*)d_in[5], buf_h, F__, 7);

  for (int layer = 0; layer < 2; ++layer) {
    const int base = 9 + 16 * layer;
    const float* A_log = (const float*)d_in[base + 0];
    const float* Dp    = (const float*)d_in[base + 1];
    const float* convb = (const float*)d_in[base + 2];
    const float* convw = (const float*)d_in[base + 3];
    const float* dt_b  = (const float*)d_in[base + 4];
    const float* dt_w  = (const float*)d_in[base + 5];
    const float* fc1_b = (const float*)d_in[base + 6];
    const float* fc1_w = (const float*)d_in[base + 7];
    const float* fc2_b = (const float*)d_in[base + 8];
    const float* fc2_w = (const float*)d_in[base + 9];
    const float* inprj = (const float*)d_in[base + 10];
    const float* ln_b  = (const float*)d_in[base + 11];
    const float* ln_w  = (const float*)d_in[base + 12];
    const float* outpr = (const float*)d_in[base + 13];
    const float* rms_w = (const float*)d_in[base + 14];
    const float* xproj = (const float*)d_in[base + 15];

    // t = rms_norm(h)
    rmsnorm_k<<<M__ / 8, blk, 0, stream>>>(buf_h, rms_w, buf_t);
    // xz = t @ in_proj^T
    gemm_wmma<<<dim3(2 * DI__ / BN, M__ / BM), blk, 0, stream>>>(
        buf_t, F__, inprj, F__, buf_xz, 2 * DI__,
        nullptr, nullptr, M__, 2 * DI__, F__, 0);
    // xm = silu(depthwise_causal_conv(xz[:, :DI]))
    dwconv_silu<<<(M__ * DI__) / 256, blk, 0, stream>>>(buf_xz, convw, convb, buf_xm);
    // xdbc = xm @ x_proj^T   (N=64)
    gemm_wmma<<<dim3(1, M__ / BM), blk, 0, stream>>>(
        buf_xm, DI__, xproj, DI__, buf_xdbc, DTR + 2 * NST,
        nullptr, nullptr, M__, DTR + 2 * NST, DI__, 0);
    // delta = softplus(xdbc[:, :32] @ dt_w^T + dt_b)  -> stored into xz[:, :DI]
    gemm_wmma<<<dim3(DI__ / BN, M__ / BM), blk, 0, stream>>>(
        buf_xdbc, DTR + 2 * NST, dt_w, DTR, buf_xz, 2 * DI__,
        dt_b, nullptr, M__, DI__, DTR, 2);
    // selective scan fused with y * silu(z); in-place over xm
    sscan_k<<<(B__ * DI__) / 256, blk, 0, stream>>>(
        buf_xm, buf_xz, buf_xdbc, buf_xz + DI__, A_log, Dp, buf_xm);
    // h = h + ygated @ out_proj^T  (residual fused in epilogue)
    gemm_wmma<<<dim3(F__ / BN, M__ / BM), blk, 0, stream>>>(
        buf_xm, DI__, outpr, DI__, buf_h, F__,
        nullptr, buf_h, M__, F__, DI__, 0);
    // t = layer_norm(h)
    layernorm_k<<<M__ / 8, blk, 0, stream>>>(buf_h, ln_w, ln_b, nullptr, buf_t);
    // xz[:, :DI] = relu(t @ fc1^T + b1)
    gemm_wmma<<<dim3(DI__ / BN, M__ / BM), blk, 0, stream>>>(
        buf_t, F__, fc1_w, F__, buf_xz, DI__,
        fc1_b, nullptr, M__, DI__, F__, 1);
    // h = xz[:, :DI] @ fc2^T + b2
    gemm_wmma<<<dim3(F__ / BN, M__ / BM), blk, 0, stream>>>(
        buf_xz, DI__, fc2_w, DI__, buf_h, F__,
        fc2_b, nullptr, M__, F__, DI__, 0);
  }

  // out = layer_norm(h) + input_seq
  layernorm_k<<<M__ / 8, blk, 0, stream>>>(buf_h, fln_w, fln_b, x_in, (float*)d_out);
}
